// AAF_GCNConv_32804960207311
// MI455X (gfx1250) — matrix operand order repaired and checked by reading, not compiled, not run on hardware
//
#include <hip/hip_runtime.h>
#include <hip/hip_bf16.h>
#include <math.h>

#define NN 50000
#define NE 800000
#define EPSV 1e-5f

typedef __attribute__((ext_vector_type(16))) __bf16 v16bf;
typedef __attribute__((ext_vector_type(8)))  float  v8f;
typedef unsigned int u32x4 __attribute__((ext_vector_type(4)));
typedef int          i32x8 __attribute__((ext_vector_type(8)));
typedef int          i32x4 __attribute__((ext_vector_type(4)));

union FragBF { v16bf v; unsigned int u[8]; };

// ---------------- utility kernels ----------------
__global__ void fill_f32(float* p, float v, int n) {
  int t = blockIdx.x * blockDim.x + threadIdx.x;
  if (t < n) p[t] = v;
}

__global__ void zero_f32(float* __restrict__ p, size_t n) {
  size_t t = (size_t)blockIdx.x * blockDim.x + threadIdx.x;
  if (t < n) p[t] = 0.f;
}

__global__ void f2bf(const float* __restrict__ in, __hip_bfloat16* __restrict__ out, size_t n) {
  size_t t = (size_t)blockIdx.x * blockDim.x + threadIdx.x;
  if (t < n) out[t] = __float2bfloat16(in[t]);
}

// W [K,N] fp32 row-major  ->  Wt [Npad,K] bf16 (column-major of B, zero-padded cols)
__global__ void wtrans(const float* __restrict__ W, __hip_bfloat16* __restrict__ Wt,
                       int K, int N, int Npad) {
  int t = blockIdx.x * blockDim.x + threadIdx.x;
  if (t >= Npad * K) return;
  int n = t / K, k = t - n * K;
  float v = (n < N) ? W[(size_t)k * N + n] : 0.f;
  Wt[t] = __float2bfloat16(v);
}

__global__ void pad_bias(const float* __restrict__ b, float* __restrict__ out, int n, int npad) {
  int t = blockIdx.x * blockDim.x + threadIdx.x;
  if (t < npad) out[t] = (t < n) ? b[t] : 0.f;
}

__global__ void deg_count(const int* __restrict__ ei, float* __restrict__ deg, int E) {
  int e = blockIdx.x * blockDim.x + threadIdx.x;
  if (e < E) atomicAdd(&deg[ei[E + e]], 1.0f);
}

__global__ void deg_finalize(const float* __restrict__ deg, float* __restrict__ isd,
                             float* __restrict__ invd, int n) {
  int t = blockIdx.x * blockDim.x + threadIdx.x;
  if (t < n) { float d = deg[t]; isd[t] = rsqrtf(d); invd[t] = 1.0f / d; }
}

// ---------------- shared epilogue helper ----------------
__device__ __forceinline__
void store_tile(v8f c, int m0, int n0, int half, int r, int N,
                float* __restrict__ outF, __hip_bfloat16* __restrict__ outB,
                const float* __restrict__ bias,
                const float* __restrict__ bnG, const float* __restrict__ bnB,
                const float* __restrict__ bnM, const float* __restrict__ bnV, int relu) {
  const int colg = n0 + r;
  float sc = 1.f, mn = 0.f, bt = 0.f;
  const bool doBN = (bnG != nullptr);
  if (doBN) {
    sc = bnG[colg] * rsqrtf(bnV[colg] + EPSV);
    mn = bnM[colg];
    bt = bnB[colg];
  }
  const float bi = bias ? bias[colg] : 0.f;
#pragma unroll
  for (int i = 0; i < 8; ++i) {
    const int rowg = m0 + i + half * 8;       // f32 C/D layout: VGPR i -> M = i + 8*half
    float v = c[i] + bi;
    if (doBN) v = (v - mn) * sc + bt;
    if (relu) v = fmaxf(v, 0.f);
    const size_t idx = (size_t)rowg * N + colg;
    if (outF) outF[idx] = v;
    if (outB) outB[idx] = __float2bfloat16(v);
  }
}

// ---------------- TDM-staged WMMA GEMM (N must be 256, M multiple of 16) ----------------
// Block = 512 threads = 16 waves; block owns one 16-row M-panel, wave w owns n-tile w.
// Wave 0 DMAs the A panel [16 x K] bf16 into LDS via the Tensor Data Mover with
// row padding (stride K/2+1 dwords) for bank-conflict-free DS reads; all 16 waves
// then run the K-loop of v_wmma_f32_16x16x32_bf16 reading A from LDS, B from L2.
__global__ __launch_bounds__(512)
void wmma_gemm_tdm(const __hip_bfloat16* __restrict__ A,
                   const __hip_bfloat16* __restrict__ Bt,   // [256,K] bf16
                   float* __restrict__ outF,
                   __hip_bfloat16* __restrict__ outB,
                   const float* __restrict__ bias,
                   const float* __restrict__ bnG, const float* __restrict__ bnB,
                   const float* __restrict__ bnM, const float* __restrict__ bnV,
                   int relu, int N, int K)
{
  __shared__ unsigned int smemA[16 * 129 + 16];   // max: 16 rows * (256/2+1) dwords
  const int lane = threadIdx.x & 31;
  const int wave = threadIdx.x >> 5;              // 0..15 -> n-tile
  const int m0 = blockIdx.x << 4;
  const int n0 = wave << 4;
  const int Kp = K >> 1;                          // dwords per row
  const int rs = Kp + 1;                          // padded LDS row stride (dwords)

  if (wave == 0) {
    // Tensor DMA descriptor (D#): 2-D tile, 16 rows x K bf16, padded into LDS.
    const unsigned long long ga = (unsigned long long)(const void*)(A + (size_t)m0 * K);
    const unsigned ldsOff = (unsigned)(unsigned long long)(const void*)smemA; // low 32b = LDS offset
    const int padInt = (K == 256) ? 6 : 5;        // pad_interval: 2^(c+1) dwords == K/2
    u32x4 g0;
    g0.x = 1u;                                    // count=1, user mode, no gather
    g0.y = ldsOff;                                // lds_addr
    g0.z = (unsigned)(ga & 0xffffffffu);          // global_addr[31:0]
    g0.w = (unsigned)(ga >> 32) | (2u << 30);     // global_addr[56:32] | type=2
    i32x8 g1;
    g1[0] = (1 << 16) | (1 << 20) | (padInt << 22); // data_size=2B, pad_enable, pad_amount=1dw
    g1[1] = (K & 0xffff) << 16;                   // tensor_dim0[15:0]
    g1[2] = (16 << 16);                           // tensor_dim0 hi | tensor_dim1=16
    g1[3] = (K << 16);                            // tensor_dim1 hi | tile_dim0=K
    g1[4] = 16;                                   // tile_dim1=16, tile_dim2=0
    g1[5] = K;                                    // tensor_dim0_stride[31:0]
    g1[6] = 0;
    g1[7] = 0;
    const i32x4 gz  = {0, 0, 0, 0};               // groups 2/3 unused (2-D tensor)
    const i32x8 gz8 = {0, 0, 0, 0, 0, 0, 0, 0};   // clang-23 6-arg form: extra group
    __builtin_amdgcn_tensor_load_to_lds(g0, g1, gz, gz, gz8, 0);
    __builtin_amdgcn_s_wait_tensorcnt(0);         // TENSORcnt -> 0 before publishing
  }
  __syncthreads();

  const int half = lane >> 4;
  const int r    = lane & 15;
  const unsigned int* __restrict__ Bpk = (const unsigned int*)Bt;
  const size_t bRow = (size_t)(n0 + r) * Kp;

  v8f c = {};
  for (int k0 = 0; k0 < K; k0 += 32) {
    FragBF a, b;
    const int kp = k0 >> 1;
#pragma unroll
    for (int i = 0; i < 8; ++i) {
      const int pi = (i < 4) ? (half * 4 + i) : (8 + half * 4 + (i - 4));
      a.u[i] = smemA[r * rs + kp + pi];           // DS read of staged A panel
      b.u[i] = Bpk[bRow + kp + pi];               // L2-resident weights
    }
    if (k0 + 32 < K)
      __builtin_prefetch(&Bpk[bRow + kp + 16], 0, 1);
    c = __builtin_amdgcn_wmma_f32_16x16x32_bf16(false, a.v, false, b.v,
                                                (short)0, c, false, false);
  }
  store_tile(c, m0, n0, half, r, N, outF, outB, bias, bnG, bnB, bnM, bnV, relu);
}

// ---------------- plain WMMA GEMM (used for the small N=48 head) ----------------
__global__ __launch_bounds__(128)
void wmma_gemm(const __hip_bfloat16* __restrict__ A,
               const __hip_bfloat16* __restrict__ Bt,
               float* __restrict__ outF,
               __hip_bfloat16* __restrict__ outB,
               const float* __restrict__ bias,
               const float* __restrict__ bnG, const float* __restrict__ bnB,
               const float* __restrict__ bnM, const float* __restrict__ bnV,
               int relu, int M, int N, int K)
{
  const int lane = threadIdx.x & 31;
  const int wave = threadIdx.x >> 5;
  const int nTilesN = N >> 4;
  const int nTiles  = (M >> 4) * nTilesN;
  const int tile = blockIdx.x * 4 + wave;       // wave-uniform
  if (tile >= nTiles) return;                   // whole-wave exit keeps EXEC all-1s
  const int m0 = (tile / nTilesN) << 4;
  const int n0 = (tile % nTilesN) << 4;

  const int half = lane >> 4;
  const int r    = lane & 15;
  const unsigned int* __restrict__ Apk = (const unsigned int*)A;
  const unsigned int* __restrict__ Bpk = (const unsigned int*)Bt;
  const int Kp = K >> 1;
  const size_t aRow = (size_t)(m0 + r) * Kp;
  const size_t bRow = (size_t)(n0 + r) * Kp;

  v8f c = {};
  for (int k0 = 0; k0 < K; k0 += 32) {
    FragBF a, b;
    const int kp = k0 >> 1;
#pragma unroll
    for (int i = 0; i < 8; ++i) {
      const int pi = (i < 4) ? (half * 4 + i) : (8 + half * 4 + (i - 4));
      a.u[i] = Apk[aRow + kp + pi];
      b.u[i] = Bpk[bRow + kp + pi];
    }
    if (k0 + 32 < K) {
      __builtin_prefetch(&Apk[aRow + kp + 16], 0, 1);
      __builtin_prefetch(&Bpk[bRow + kp + 16], 0, 1);
    }
    c = __builtin_amdgcn_wmma_f32_16x16x32_bf16(false, a.v, false, b.v,
                                                (short)0, c, false, false);
  }
  store_tile(c, m0, n0, half, r, N, outF, outB, bias, bnG, bnB, bnM, bnV, relu);
}

// ---------------- edge scatter: agg[dst] += hw[src] * isd[src]*isd[dst] ----------------
__global__ __launch_bounds__(256)
void edge_scatter(const int* __restrict__ ei, const float* __restrict__ hw,
                  const float* __restrict__ isd, float* __restrict__ agg, int E)
{
  const int tid = blockIdx.x * blockDim.x + threadIdx.x;   // E*64 threads
  const int e = tid >> 6;
  if (e >= E) return;
  const int g = tid & 63;
  const int s = ei[e];
  const int d = ei[E + e];
  const float w = isd[s] * isd[d];
  const float4 m = *(const float4*)(hw + (size_t)s * 256 + g * 4);
  float* dp = agg + (size_t)d * 256 + g * 4;
  atomicAdd(dp + 0, m.x * w);
  atomicAdd(dp + 1, m.y * w);
  atomicAdd(dp + 2, m.z * w);
  atomicAdd(dp + 3, m.w * w);
}

// ---------------- self-loop + bias + BN + ReLU -> bf16 for next WMMA layer ----------------
__global__ __launch_bounds__(256)
void combine_bn_relu(const float* __restrict__ agg, const float* __restrict__ hw,
                     const float* __restrict__ invd, const float* __restrict__ bias,
                     const float* __restrict__ g, const float* __restrict__ be,
                     const float* __restrict__ mn, const float* __restrict__ vr,
                     __hip_bfloat16* __restrict__ hbf)
{
  const size_t t = (size_t)blockIdx.x * blockDim.x + threadIdx.x;
  if (t >= (size_t)NN * 256) return;
  const int n = (int)(t >> 8);
  const int f = (int)(t & 255);
  float v = agg[t] + hw[t] * invd[n] + bias[f];
  const float s = g[f] * rsqrtf(vr[f] + EPSV);
  v = (v - mn[f]) * s + be[f];
  v = fmaxf(v, 0.f);
  hbf[t] = __float2bfloat16(v);
}

// ---------------- log_softmax over 40 classes (stride-48 input), wave per row ----------
__global__ __launch_bounds__(256)
void log_softmax40(const float* __restrict__ logits, float* __restrict__ out, int M)
{
  const int lane = threadIdx.x & 31;
  const int row  = blockIdx.x * 8 + (threadIdx.x >> 5);
  if (row >= M) return;
  const float* rp = logits + (size_t)row * 48;
  const float a = rp[lane];
  const float b = (lane < 8) ? rp[32 + lane] : -3.402823466e38f;
  float m = fmaxf(a, b);
#pragma unroll
  for (int off = 16; off > 0; off >>= 1) m = fmaxf(m, __shfl_xor(m, off, 32));
  float s = __expf(a - m) + ((lane < 8) ? __expf(b - m) : 0.f);
#pragma unroll
  for (int off = 16; off > 0; off >>= 1) s += __shfl_xor(s, off, 32);
  const float lse = m + __logf(s);
  float* op = out + (size_t)row * 40;
  op[lane] = a - lse;
  if (lane < 8) op[32 + lane] = b - lse;
}

// ---------------- driver ----------------
extern "C" void kernel_launch(void* const* d_in, const int* in_sizes, int n_in,
                              void* d_out, int out_size, void* d_ws, size_t ws_size,
                              hipStream_t stream) {
  const float* x       = (const float*)d_in[0];
  const int*   ei      = (const int*)d_in[1];
  const float* w_pre   = (const float*)d_in[2];
  const float* b_pre   = (const float*)d_in[3];
  const float* w_conv  = (const float*)d_in[4];
  const float* b_conv  = (const float*)d_in[5];
  const float* w_post1 = (const float*)d_in[6];
  const float* b_post1 = (const float*)d_in[7];
  const float* w_post2 = (const float*)d_in[8];
  const float* b_post2 = (const float*)d_in[9];
  const float* bn_g    = (const float*)d_in[10];
  const float* bn_b    = (const float*)d_in[11];
  const float* bn_m    = (const float*)d_in[12];
  const float* bn_v    = (const float*)d_in[13];
  float* out = (float*)d_out;

  unsigned char* wsp = (unsigned char*)d_ws;
  auto take = [&](size_t bytes) -> void* {
    void* p = (void*)wsp;
    wsp += (bytes + 255) & ~(size_t)255;
    return p;
  };
  float* deg   = (float*)take(NN * 4);
  float* isd   = (float*)take(NN * 4);
  float* invd  = (float*)take(NN * 4);
  __hip_bfloat16* xbf      = (__hip_bfloat16*)take((size_t)NN * 128 * 2);
  __hip_bfloat16* wpre_t   = (__hip_bfloat16*)take(256 * 128 * 2);
  __hip_bfloat16* wconv_t  = (__hip_bfloat16*)take(3 * 256 * 256 * 2);
  __hip_bfloat16* wpost1_t = (__hip_bfloat16*)take(256 * 256 * 2);
  __hip_bfloat16* wpost2_t = (__hip_bfloat16*)take(48 * 256 * 2);
  float* bias48 = (float*)take(48 * 4);
  __hip_bfloat16* hbf  = (__hip_bfloat16*)take((size_t)NN * 256 * 2);
  __hip_bfloat16* hbf2 = (__hip_bfloat16*)take((size_t)NN * 256 * 2);
  float* hw     = (float*)take((size_t)NN * 256 * 4);
  float* agg    = (float*)take((size_t)NN * 256 * 4);
  float* logits = (float*)take((size_t)NN * 48 * 4);

  // ---- degree / norm coefficients ----
  fill_f32<<<(NN + 255) / 256, 256, 0, stream>>>(deg, 1.0f, NN);
  deg_count<<<(NE + 255) / 256, 256, 0, stream>>>(ei, deg, NE);
  deg_finalize<<<(NN + 255) / 256, 256, 0, stream>>>(deg, isd, invd, NN);

  // ---- bf16 conversions / weight transposes ----
  f2bf<<<(int)(((size_t)NN * 128 + 255) / 256), 256, 0, stream>>>(x, xbf, (size_t)NN * 128);
  wtrans<<<(256 * 128 + 255) / 256, 256, 0, stream>>>(w_pre, wpre_t, 128, 256, 256);
  for (int i = 0; i < 3; ++i)
    wtrans<<<(256 * 256 + 255) / 256, 256, 0, stream>>>(w_conv + (size_t)i * 256 * 256,
                                                        wconv_t + (size_t)i * 256 * 256,
                                                        256, 256, 256);
  wtrans<<<(256 * 256 + 255) / 256, 256, 0, stream>>>(w_post1, wpost1_t, 256, 256, 256);
  wtrans<<<(48 * 256 + 255) / 256, 256, 0, stream>>>(w_post2, wpost2_t, 256, 40, 48);
  pad_bias<<<1, 64, 0, stream>>>(b_post2, bias48, 40, 48);

  // ---- pre-layer: h = relu(x @ w_pre + b_pre) -> bf16 (TDM-staged, 3125 M-panels) ----
  wmma_gemm_tdm<<<NN / 16, 512, 0, stream>>>(xbf, wpre_t, nullptr, hbf, b_pre,
                                             nullptr, nullptr, nullptr, nullptr,
                                             1, 256, 128);

  // ---- 3 GCN conv layers ----
  for (int i = 0; i < 3; ++i) {
    wmma_gemm_tdm<<<NN / 16, 512, 0, stream>>>(hbf, wconv_t + (size_t)i * 256 * 256,
                                               hw, nullptr, nullptr,
                                               nullptr, nullptr, nullptr, nullptr,
                                               0, 256, 256);
    zero_f32<<<(int)(((size_t)NN * 256 + 255) / 256), 256, 0, stream>>>(agg, (size_t)NN * 256);
    edge_scatter<<<(NE * 64) / 256, 256, 0, stream>>>(ei, hw, isd, agg, NE);
    combine_bn_relu<<<(int)(((size_t)NN * 256 + 255) / 256), 256, 0, stream>>>(
        agg, hw, invd, b_conv + (size_t)i * 256,
        bn_g + (size_t)i * 256, bn_b + (size_t)i * 256,
        bn_m + (size_t)i * 256, bn_v + (size_t)i * 256, hbf);
  }

  // ---- post1: relu(bn3(h @ w_post1 + b_post1)) -> bf16 (separate buffer, no RAW) ----
  wmma_gemm_tdm<<<NN / 16, 512, 0, stream>>>(hbf, wpost1_t, nullptr, hbf2, b_post1,
                                             bn_g + 3 * 256, bn_b + 3 * 256,
                                             bn_m + 3 * 256, bn_v + 3 * 256,
                                             1, 256, 256);

  // ---- post2: logits[50000,48] = h @ w_post2(pad48) + bias48 ----
  wmma_gemm<<<2344, 128, 0, stream>>>(hbf2, wpost2_t, logits, nullptr, bias48,
                                      nullptr, nullptr, nullptr, nullptr,
                                      0, NN, 48, 256);

  // ---- log_softmax over 40 classes ----
  log_softmax40<<<(NN + 7) / 8, 256, 0, stream>>>(logits, out, NN);
}
